// TIBiLSTM_84207128805842
// MI455X (gfx1250) — compile-verified, matrix operand order checked by reading, not run on hardware
//
#include <hip/hip_runtime.h>
#include <stdint.h>

#define T_SEQ   16384
#define TOK_D   100
#define POS_D   32
#define LEM_D   64
#define WV_D    100
#define EMB_RAW 296
#define EMB_PAD 320
#define IN_D    512
#define HID     512
#define G4      2048   // 4*HID
#define PROJ    256

typedef __attribute__((ext_vector_type(16))) __bf16 v16bf;
typedef __attribute__((ext_vector_type(8)))  float  v8f;
typedef int v4i __attribute__((vector_size(16)));   // matches builtin's expected pointee

__device__ __forceinline__ uint16_t f2bf(float f) {
  uint32_t u = __builtin_bit_cast(uint32_t, f);
  u += 0x7FFFu + ((u >> 16) & 1u);
  return (uint16_t)(u >> 16);
}
__device__ __forceinline__ float bf2f(uint16_t h) {
  uint32_t u = ((uint32_t)h) << 16;
  return __builtin_bit_cast(float, u);
}
__device__ __forceinline__ float sigm(float x) { return 1.0f / (1.0f + __expf(-x)); }

// Async global->LDS 16-byte copy (CDNA5 GLOBAL_LOAD_ASYNC_TO_LDS_B128, ASYNCcnt-tracked).
__device__ __forceinline__ void async_copy_16B(const uint16_t* gp, uint16_t* lp) {
#if __has_builtin(__builtin_amdgcn_global_load_async_to_lds_b128)
  __builtin_amdgcn_global_load_async_to_lds_b128(
      (__attribute__((address_space(1))) v4i*)gp,
      (__attribute__((address_space(3))) v4i*)lp, 0, 0);
#else
  uint32_t loff = (uint32_t)(uintptr_t)(__attribute__((address_space(3))) uint16_t*)lp;
  asm volatile("global_load_async_to_lds_b128 %0, %1, off"
               :: "v"(loff), "v"(gp) : "memory");
#endif
}
__device__ __forceinline__ void wait_async0() {
  asm volatile("s_wait_asynccnt 0" ::: "memory");
}

// ---------------- weight conversion ----------------
__global__ void k_f32_to_bf16(const float* __restrict__ src, uint16_t* __restrict__ dst, int n) {
  for (int i = blockIdx.x * blockDim.x + threadIdx.x; i < n; i += gridDim.x * blockDim.x)
    dst[i] = f2bf(src[i]);
}

__global__ void k_pad_linW(const float* __restrict__ src, uint16_t* __restrict__ dst) {
  int i = blockIdx.x * blockDim.x + threadIdx.x;  // over 512*320
  if (i >= IN_D * EMB_PAD) return;
  int row = i / EMB_PAD, col = i % EMB_PAD;
  dst[i] = (col < EMB_RAW) ? f2bf(src[row * EMB_RAW + col]) : (uint16_t)0;
}

// ---------------- embedding gather (bf16, K padded to 320) ----------------
__global__ void k_gather(const int* __restrict__ tokens, const int* __restrict__ postags,
                         const int* __restrict__ lemmas,
                         const float* __restrict__ emb_token, const float* __restrict__ emb_pos,
                         const float* __restrict__ emb_lem, const float* __restrict__ word_vec,
                         uint16_t* __restrict__ out) {
  int t = blockIdx.x;
  int c = threadIdx.x;  // 0..319
  int tok = tokens[t];
  float v;
  if (c < TOK_D)                       v = emb_token[(size_t)tok * TOK_D + c];
  else if (c < TOK_D + POS_D)          v = emb_pos[(size_t)postags[t] * POS_D + (c - TOK_D)];
  else if (c < TOK_D + POS_D + LEM_D)  v = emb_lem[(size_t)lemmas[t] * LEM_D + (c - TOK_D - POS_D)];
  else if (c < EMB_RAW)                v = word_vec[(size_t)tok * WV_D + (c - TOK_D - POS_D - LEM_D)];
  else                                 v = 0.0f;
  out[(size_t)t * EMB_PAD + c] = f2bf(v);
}

// ---------------- WMMA bf16 GEMM: out[M,N] = act(A[M,K] @ W[N,K]^T + b1 + b2) ----------------
// A, W row-major bf16; out bf16. One 16x16 tile per wave; 8 waves/block -> 16x128 per block.
// A panel (16xK, shared by all 8 waves) is staged once into LDS via async-to-LDS copies;
// B streams from global straight into the WMMA pipeline.
// Operand packing per CDNA5 16-bit layouts: lane<16 holds K {0..7,16..23}, lane>=16 holds {8..15,24..31}.
union ABu { v16bf v; uint4 q[2]; };

#define A_STR_MAX (512 + 8)

__global__ void k_gemm_bf16(const uint16_t* __restrict__ A, const uint16_t* __restrict__ W,
                            const float* __restrict__ bias1, const float* __restrict__ bias2,
                            uint16_t* __restrict__ out, int M, int N, int K,
                            int relu, int revA) {
  __shared__ uint16_t aPanel[16 * A_STR_MAX];
  const int ASTR = K + 8;                       // padded LDS row stride (bank spread)
  const int tid = threadIdx.x;
  const int m0 = blockIdx.x * 16;

  // async stage: 16 rows x K bf16 -> LDS (revA maps time-reversed rows for the backward LSTM)
  {
    const int chunksPerRow = K >> 3;            // 16B chunks per row
    const int nChunks = chunksPerRow << 4;      // 16 rows
    for (int idx = tid; idx < nChunks; idx += 256) {
      int r = idx / chunksPerRow;
      int cc = idx - r * chunksPerRow;
      int arow = m0 + r;
      if (revA) arow = M - 1 - arow;
      async_copy_16B(A + (size_t)arow * K + (cc << 3), aPanel + r * ASTR + (cc << 3));
    }
  }
  wait_async0();
  __syncthreads();

  const int wave = tid >> 5;
  const int lane = tid & 31;
  const int n0 = (blockIdx.y * 8 + wave) * 16;
  const int brow = n0 + (lane & 15);
  const int kb = (lane >> 4) * 8;

  v8f c = {};
  const uint16_t* apb = aPanel + (lane & 15) * ASTR + kb;
  const uint16_t* bpb = W + (size_t)brow * K + kb;
  for (int kc = 0; kc < K; kc += 32) {
    ABu a, b;
    a.q[0] = *(const uint4*)(apb + kc);         // ds_load_b128 from staged panel
    a.q[1] = *(const uint4*)(apb + kc + 16);
    b.q[0] = *(const uint4*)(bpb + kc);         // global_load_b128, pipelined with WMMA
    b.q[1] = *(const uint4*)(bpb + kc + 16);
    c = __builtin_amdgcn_wmma_f32_16x16x32_bf16(false, a.v, false, b.v, (short)0, c, false, false);
  }

  int n = n0 + (lane & 15);
  float bsum = (bias1 ? bias1[n] : 0.0f) + (bias2 ? bias2[n] : 0.0f);
#pragma unroll
  for (int v = 0; v < 8; ++v) {
    int m = m0 + v + ((lane >> 4) << 3);        // VGPR v: M=v (lanes 0-15), M=v+8 (lanes 16-31)
    float val = c[v] + bsum;
    if (relu) val = fmaxf(val, 0.0f);
    out[(size_t)m * N + n] = f2bf(val);
  }
}

// ---------------- recurrence: 8 WGs per direction, LDS-resident recurrent weights ----------------
#define WHH_LD 260   // padded row stride (uint16) -> bank-conflict-free wave32 access
#define WHR_LD 66
#define RECUR_SMEM (256*WHH_LD*2 + 256*WHR_LD*2 + (256+256+64+64)*4)

__global__ void k_zero_ctr(int* ctr) {
  if (threadIdx.x < 2) ctr[threadIdx.x] = 0;
}

__global__ void k_recur(const uint16_t* __restrict__ whhF, const uint16_t* __restrict__ whhB,
                        const uint16_t* __restrict__ whrF, const uint16_t* __restrict__ whrB,
                        const uint16_t* __restrict__ gxF, const uint16_t* __restrict__ gxB,
                        float* __restrict__ hF, float* __restrict__ hBrev,
                        float* __restrict__ partials, int* __restrict__ ctr) {
  extern __shared__ char smem[];
  uint16_t* whh_s = (uint16_t*)smem;                       // [256][WHH_LD]
  uint16_t* whr_s = whh_s + 256 * WHH_LD;                  // [256][WHR_LD]
  float* h_s  = (float*)(whr_s + 256 * WHR_LD);            // [256]
  float* g_s  = h_s + 256;                                 // [256]
  float* c_s  = g_s + 256;                                 // [64]
  float* ot_s = c_s + 64;                                  // [64]

  const int dir = blockIdx.x >> 3;          // 0 = forward, 1 = backward
  const int w   = blockIdx.x & 7;           // workgroup slice within direction
  const int cb  = w * 64;                   // owned cell indices [cb, cb+64)
  const int tid = threadIdx.x;              // 256 threads

  const uint16_t* whh = dir ? whhB : whhF;
  const uint16_t* whr = dir ? whrB : whrF;
  const uint16_t* gx  = dir ? gxB  : gxF;
  float* hs           = dir ? hBrev : hF;

  // stage recurrent-weight slices into LDS (stay resident for all 16384 steps)
  for (int idx = tid; idx < 256 * 256; idx += 256) {
    int r = idx >> 8, col = idx & 255;                     // r: 4 gates x 64 cells
    int grow = (r >> 6) * HID + cb + (r & 63);
    whh_s[r * WHH_LD + col] = whh[(size_t)grow * PROJ + col];
  }
  for (int idx = tid; idx < 256 * 64; idx += 256) {
    int p = idx >> 6, k = idx & 63;
    whr_s[p * WHR_LD + k] = whr[(size_t)p * HID + cb + k];
  }
  h_s[tid] = 0.0f;
  if (tid < 64) c_s[tid] = 0.0f;
  __syncthreads();

  for (int t = 0; t < T_SEQ; ++t) {
    // g[r] = gx[t][grow] + dot(Whh_row, h)
    {
      int r = tid;
      int grow = (r >> 6) * HID + cb + (r & 63);
      float acc = bf2f(gx[(size_t)t * G4 + grow]);
      const uint32_t* wrow = (const uint32_t*)(whh_s + (size_t)r * WHH_LD);
#pragma unroll 8
      for (int k2 = 0; k2 < 128; ++k2) {
        uint32_t pk = wrow[k2];
        acc += bf2f((uint16_t)(pk & 0xFFFF)) * h_s[2 * k2];
        acc += bf2f((uint16_t)(pk >> 16))    * h_s[2 * k2 + 1];
      }
      g_s[r] = acc;
    }
    __syncthreads();

    // gates + cell update for the 64 owned cells
    if (tid < 64) {
      float i  = sigm(g_s[tid]);
      float f  = sigm(g_s[64 + tid]);
      float gg = tanhf(g_s[128 + tid]);
      float o  = sigm(g_s[192 + tid]);
      float cc = f * c_s[tid] + i * gg;
      c_s[tid]  = cc;
      ot_s[tid] = o * tanhf(cc);
    }
    __syncthreads();

    // partial projection: h_part[p] = sum_{k in slice} ot[k] * Whr[p][k]
    {
      int p = tid;
      float acc = 0.0f;
      const uint32_t* wr = (const uint32_t*)(whr_s + (size_t)p * WHR_LD);
#pragma unroll
      for (int k2 = 0; k2 < 32; ++k2) {
        uint32_t pk = wr[k2];
        acc += bf2f((uint16_t)(pk & 0xFFFF)) * ot_s[2 * k2];
        acc += bf2f((uint16_t)(pk >> 16))    * ot_s[2 * k2 + 1];
      }
      partials[(((size_t)(t & 1) * 2 + dir) * 8 + w) * 256 + p] = acc;
    }
    __syncthreads();

    // per-direction grid barrier (monotonic counter; partials double-buffered by t parity)
    if (tid == 0) {
      __hip_atomic_fetch_add(&ctr[dir], 1, __ATOMIC_RELEASE, __HIP_MEMORY_SCOPE_AGENT);
      int target = 8 * (t + 1);
      while (__hip_atomic_load(&ctr[dir], __ATOMIC_ACQUIRE, __HIP_MEMORY_SCOPE_AGENT) < target)
        __builtin_amdgcn_s_sleep(2);
    }
    __syncthreads();

    // deterministic reduction of the 8 partials -> new h
    {
      int p = tid;
      const float* pb = partials + ((size_t)(t & 1) * 2 + dir) * 8 * 256 + p;
      float s = 0.0f;
#pragma unroll
      for (int w2 = 0; w2 < 8; ++w2) s += pb[(size_t)w2 * 256];
      h_s[p] = s;
      if (w == 0) hs[(size_t)t * PROJ + p] = s;
    }
    __syncthreads();
  }
}

// ---------------- final FC: scores[t] = [hf[t], hb[t]] @ fcW.T + fcb ----------------
__global__ void k_fc(const float* __restrict__ hF, const float* __restrict__ hBrev,
                     const float* __restrict__ fcW, const float* __restrict__ fcb,
                     float* __restrict__ out) {
  int t = blockIdx.x * blockDim.x + threadIdx.x;
  if (t >= T_SEQ) return;
  const float* hf = hF + (size_t)t * PROJ;
  const float* hb = hBrev + (size_t)(T_SEQ - 1 - t) * PROJ;
#pragma unroll
  for (int o = 0; o < 2; ++o) {
    const float* wv = fcW + (size_t)o * 2 * PROJ;
    float acc = fcb[o];
    for (int p = 0; p < PROJ; ++p) acc += hf[p] * wv[p];
    for (int p = 0; p < PROJ; ++p) acc += hb[p] * wv[PROJ + p];
    out[(size_t)t * 2 + o] = acc;
  }
}

// ---------------- host launch ----------------
extern "C" void kernel_launch(void* const* d_in, const int* in_sizes, int n_in,
                              void* d_out, int out_size, void* d_ws, size_t ws_size,
                              hipStream_t stream) {
  const int*   tokens   = (const int*)d_in[0];
  const int*   postags  = (const int*)d_in[1];
  const int*   lemmas   = (const int*)d_in[2];
  const float* embTok   = (const float*)d_in[3];
  const float* embPos   = (const float*)d_in[4];
  const float* embLem   = (const float*)d_in[5];
  const float* wordVec  = (const float*)d_in[6];
  const float* linW     = (const float*)d_in[7];
  const float* linb     = (const float*)d_in[8];
  const float* WihF     = (const float*)d_in[9];
  const float* WhhF     = (const float*)d_in[10];
  const float* bihF     = (const float*)d_in[11];
  const float* bhhF     = (const float*)d_in[12];
  const float* WhrF     = (const float*)d_in[13];
  const float* WihB     = (const float*)d_in[14];
  const float* WhhB     = (const float*)d_in[15];
  const float* bihB     = (const float*)d_in[16];
  const float* bhhB     = (const float*)d_in[17];
  const float* WhrB     = (const float*)d_in[18];
  const float* fcW      = (const float*)d_in[19];
  const float* fcb      = (const float*)d_in[20];
  float* out = (float*)d_out;

  // workspace carve (256B aligned)
  char* ws = (char*)d_ws;
  size_t off = 0;
  auto carve = [&](size_t bytes) -> char* {
    char* p = ws + off;
    off = (off + bytes + 255) & ~(size_t)255;
    return p;
  };
  uint16_t* xEmb   = (uint16_t*)carve((size_t)T_SEQ * EMB_PAD * 2);
  uint16_t* linWb  = (uint16_t*)carve((size_t)IN_D * EMB_PAD * 2);
  uint16_t* xb     = (uint16_t*)carve((size_t)T_SEQ * IN_D * 2);
  uint16_t* WihFb  = (uint16_t*)carve((size_t)G4 * IN_D * 2);
  uint16_t* WihBb  = (uint16_t*)carve((size_t)G4 * IN_D * 2);
  uint16_t* WhhFb  = (uint16_t*)carve((size_t)G4 * PROJ * 2);
  uint16_t* WhhBb  = (uint16_t*)carve((size_t)G4 * PROJ * 2);
  uint16_t* WhrFb  = (uint16_t*)carve((size_t)PROJ * HID * 2);
  uint16_t* WhrBb  = (uint16_t*)carve((size_t)PROJ * HID * 2);
  uint16_t* gxF    = (uint16_t*)carve((size_t)T_SEQ * G4 * 2);
  uint16_t* gxB    = (uint16_t*)carve((size_t)T_SEQ * G4 * 2);
  float*    hF     = (float*)carve((size_t)T_SEQ * PROJ * 4);
  float*    hBrev  = (float*)carve((size_t)T_SEQ * PROJ * 4);
  float*    parts  = (float*)carve((size_t)2 * 2 * 8 * 256 * 4);
  int*      ctr    = (int*)carve(64);

  // weight conversion (f32 -> bf16)
  auto cvt = [&](const float* s, uint16_t* d, int n) {
    k_f32_to_bf16<<<(n + 255) / 256, 256, 0, stream>>>(s, d, n);
  };
  k_pad_linW<<<(IN_D * EMB_PAD + 255) / 256, 256, 0, stream>>>(linW, linWb);
  cvt(WihF, WihFb, G4 * IN_D);
  cvt(WihB, WihBb, G4 * IN_D);
  cvt(WhhF, WhhFb, G4 * PROJ);
  cvt(WhhB, WhhBb, G4 * PROJ);
  cvt(WhrF, WhrFb, PROJ * HID);
  cvt(WhrB, WhrBb, PROJ * HID);

  // embedding gather
  k_gather<<<T_SEQ, EMB_PAD, 0, stream>>>(tokens, postags, lemmas, embTok, embPos, embLem,
                                          wordVec, xEmb);

  // x = relu(emb @ linW.T + linb)            [T,512], K=320
  k_gemm_bf16<<<dim3(T_SEQ / 16, IN_D / 128), 256, 0, stream>>>(
      xEmb, linWb, linb, nullptr, xb, T_SEQ, IN_D, EMB_PAD, /*relu=*/1, /*rev=*/0);

  // gx_f = x @ WihF.T + (bihF+bhhF)          [T,2048], K=512
  k_gemm_bf16<<<dim3(T_SEQ / 16, G4 / 128), 256, 0, stream>>>(
      xb, WihFb, bihF, bhhF, gxF, T_SEQ, G4, IN_D, 0, 0);

  // gx_b = x[::-1] @ WihB.T + (bihB+bhhB)    reversed-row A
  k_gemm_bf16<<<dim3(T_SEQ / 16, G4 / 128), 256, 0, stream>>>(
      xb, WihBb, bihB, bhhB, gxB, T_SEQ, G4, IN_D, 0, 1);

  // recurrence: 16 persistent workgroups (8 per direction), big dynamic LDS
  k_zero_ctr<<<1, 32, 0, stream>>>(ctr);
  (void)hipFuncSetAttribute((const void*)k_recur,
                            hipFuncAttributeMaxDynamicSharedMemorySize, RECUR_SMEM);
  k_recur<<<16, 256, RECUR_SMEM, stream>>>(WhhFb, WhhBb, WhrFb, WhrBb, gxF, gxB,
                                           hF, hBrev, parts, ctr);

  // final FC
  k_fc<<<(T_SEQ + 255) / 256, 256, 0, stream>>>(hF, hBrev, fcW, fcb, out);
}
